// MultiHeadAttention_61907658605180
// MI455X (gfx1250) — compile-verified, matrix operand order checked by reading, not run on hardware
//
#include <hip/hip_runtime.h>

typedef __attribute__((ext_vector_type(16))) _Float16 v16h;
typedef __attribute__((ext_vector_type(8)))  float    v8f;
typedef int v4i __attribute__((vector_size(16)));

#define WMMA_F32_F16(A, B, C) \
  __builtin_amdgcn_wmma_f32_16x16x32_f16(false, (A), false, (B), (short)0, (C), false, false)

// ---- gfx1250 async global->LDS copy (ASYNCcnt-tracked), with sync fallback --
#if defined(__has_builtin)
#  if __has_builtin(__builtin_amdgcn_global_load_async_to_lds_b128)
#    define HAVE_ASYNC_LDS 1
#  endif
#endif
#ifndef HAVE_ASYNC_LDS
#  define HAVE_ASYNC_LDS 0
#endif

__device__ __forceinline__ void async_copy16(const _Float16* g, _Float16* l) {
#if HAVE_ASYNC_LDS
  __builtin_amdgcn_global_load_async_to_lds_b128(
      (__attribute__((address_space(1))) v4i*)g,
      (__attribute__((address_space(3))) v4i*)l, 0, 0);
#else
  *(int4*)l = *(const int4*)g;
#endif
}

__device__ __forceinline__ void async_join() {
#if HAVE_ASYNC_LDS
  asm volatile("s_wait_asynccnt 0" ::: "memory");
#endif
}

// Load a 16x32 f16 fragment from row-major memory (row stride = ld elements).
// Serves WMMA A, and B when the source is stored as B^T ([N][K] row-major):
// the CDNA5 A/B VGPR layouts are mirror images (lane = M or N, elements = K).
// 16-bit A layout: lane l: M=l%16, element pair v holds
// K = (v<4?0:16) + (l/16)*8 + (v%4)*2 + {0,1}.
__device__ __forceinline__ v16h load_frag(const _Float16* __restrict__ base,
                                          long ld, int lane) {
  const int row = lane & 15;
  const int g = (lane >> 4) & 1;
  const _Float16* p = base + (long)row * ld;
  v16h f;
#pragma unroll
  for (int v = 0; v < 8; ++v) {
    const int k = ((v & 4) << 2) | (g << 3) | ((v & 3) << 1);
    f[2 * v]     = p[k];
    f[2 * v + 1] = p[k + 1];
  }
  return f;
}

// ---------------------------------------------------------------------------
// f32 -> f16 conversion
__global__ __launch_bounds__(256) void cvt_f16_kernel(
    const float* __restrict__ in, _Float16* __restrict__ out, int n) {
  int idx = blockIdx.x * 256 + threadIdx.x;
  if (idx < n) out[idx] = (_Float16)in[idx];
}

// in: [K][N] f32 row-major  ->  out: [N][K] f16 row-major (transpose+convert)
__global__ __launch_bounds__(256) void transpose_cvt_kernel(
    const float* __restrict__ in, _Float16* __restrict__ out, int K, int N) {
  int idx = blockIdx.x * 256 + threadIdx.x;
  if (idx >= K * N) return;
  int n = idx / K;
  int k = idx - n * K;
  out[idx] = (_Float16)in[(long)k * N + n];
}

// ---------------------------------------------------------------------------
// QKV projection: C[8192][3072] = Xh[8192][1024] @ WqkvT^T (+bias), scattered
// into Q [64][2048][64], K [64][2048][64], V^T [64][64][2048] (f16).
// Block tile 128(M)x128(N); A/B K-slices (128x32 f16) double-buffered in LDS
// via async copies; each of 8 waves computes 32x64 (8 WMMAs / K-step).
__global__ __launch_bounds__(256) void qkv_gemm_kernel(
    const _Float16* __restrict__ A,   // [8192][1024]
    const _Float16* __restrict__ BT,  // [3072][1024]
    const float* __restrict__ bias,   // [3072]
    _Float16* __restrict__ Qb, _Float16* __restrict__ Kb,
    _Float16* __restrict__ Vt) {
  __shared__ __align__(16) _Float16 Abuf[2][128 * 32];
  __shared__ __align__(16) _Float16 Bbuf[2][128 * 32];
  const int tid = threadIdx.x, lane = tid & 31, wave = tid >> 5;
  const int wm = wave & 3, wn = wave >> 2;
  const long Arow0 = (long)blockIdx.y * 128;
  const long Brow0 = (long)blockIdx.x * 128;

  auto stage = [&](int buf, int k) {
#pragma unroll
    for (int j = 0; j < 2; ++j) {
      const int c = tid + 256 * j;            // 512 16B chunks per matrix
      const int row = c >> 2, seg = (c & 3) << 3;
      async_copy16(A + (Arow0 + row) * 1024 + k + seg, &Abuf[buf][row * 32 + seg]);
      async_copy16(BT + (Brow0 + row) * 1024 + k + seg, &Bbuf[buf][row * 32 + seg]);
    }
  };

  v8f acc[2][4] = {};
  stage(0, 0);
  async_join();
  __syncthreads();
  for (int ib = 0; ib < 32; ++ib) {
    const int cur = ib & 1;
    if (ib + 1 < 32) stage(cur ^ 1, (ib + 1) * 32);

    v16h a0 = load_frag(&Abuf[cur][(wm * 32) * 32], 32, lane);
    v16h a1 = load_frag(&Abuf[cur][(wm * 32 + 16) * 32], 32, lane);
#pragma unroll
    for (int j = 0; j < 4; ++j) {
      v16h b = load_frag(&Bbuf[cur][(wn * 64 + j * 16) * 32], 32, lane);
      acc[0][j] = WMMA_F32_F16(a0, b, acc[0][j]);
      acc[1][j] = WMMA_F32_F16(a1, b, acc[1][j]);
    }
    async_join();
    __syncthreads();
  }

  const int m0 = (int)Arow0 + wm * 32;
  const int n0 = (int)Brow0 + wn * 64;
  const int g = lane >> 4, cn = lane & 15;
#pragma unroll
  for (int i = 0; i < 2; ++i)
#pragma unroll
    for (int j = 0; j < 4; ++j)
#pragma unroll
      for (int r = 0; r < 8; ++r) {
        const int gm = m0 + i * 16 + (g << 3) + r;   // token index
        const int gn = n0 + j * 16 + cn;             // qkv column
        const float val = acc[i][j][r] + bias[gn];
        const int b = gm >> 11, s = gm & 2047;
        const int sec = gn >> 10, nn = gn & 1023;    // 0=Q 1=K 2=V
        const int h = nn >> 6, hd = nn & 63;
        const long bh = (long)(b * 16 + h);
        if (sec == 0)      Qb[(bh * 2048 + s) * 64 + hd] = (_Float16)val;
        else if (sec == 1) Kb[(bh * 2048 + s) * 64 + hd] = (_Float16)val;
        else               Vt[(bh * 64 + hd) * 2048 + s] = (_Float16)val;
      }
}

// ---------------------------------------------------------------------------
// Flash attention: one workgroup = 8 waves on the same (b,h), 8 adjacent
// 16-query tiles (128 queries). 32-key K/V tiles are staged once into LDS
// (async, double-buffered) and shared by all 8 waves. Per key block per wave:
// 4 WMMAs for scores, shuffle-reduced online softmax, P->LDS layout bounce,
// 4 WMMAs for PV.
__global__ __launch_bounds__(256) void flash_attn_kernel(
    const _Float16* __restrict__ Qb, const _Float16* __restrict__ Kb,
    const _Float16* __restrict__ Vt, _Float16* __restrict__ attn) {
  __shared__ __align__(16) _Float16 Ktile[2][32 * 64];  // [key][hd]
  __shared__ __align__(16) _Float16 Vtile[2][64 * 32];  // [hd][key]
  __shared__ __align__(16) _Float16 pbuf[8][16 * 32];

  const int tid = threadIdx.x;
  const int lane = tid & 31;
  const int wave = tid >> 5;
  const int bh = blockIdx.x >> 4;     // batch*head (0..63)
  const int qblk = blockIdx.x & 15;   // 128-query block

  const _Float16* Qh = Qb + (long)bh * (2048 * 64) + (long)(qblk * 128 + wave * 16) * 64;
  const _Float16* Kh = Kb + (long)bh * (2048 * 64);
  const _Float16* Vh = Vt + (long)bh * (64 * 2048);

  auto stageK = [&](int buf, int kb) {  // 32 x 64 halves = 256 x 16B
    const int row = tid >> 3, seg = (tid & 7) << 3;
    async_copy16(Kh + (long)(kb + row) * 64 + seg, &Ktile[buf][row * 64 + seg]);
  };
  auto stageV = [&](int buf, int kb) {  // 64 x 32 halves = 256 x 16B
    const int row = tid >> 2, seg = (tid & 3) << 3;
    async_copy16(Vh + (long)row * 2048 + kb + seg, &Vtile[buf][row * 32 + seg]);
  };

  const v16h qa0 = load_frag(Qh, 64, lane);        // hd 0..31
  const v16h qa1 = load_frag(Qh + 32, 64, lane);   // hd 32..63

  v8f o[4] = {};
  float mrow[8], lrow[8];
#pragma unroll
  for (int r = 0; r < 8; ++r) { mrow[r] = -3.0e38f; lrow[r] = 0.0f; }

  const int g = lane >> 4, cn = lane & 15;
  _Float16* pw = &pbuf[wave][0];

  stageK(0, 0);
  stageV(0, 0);
  async_join();
  __syncthreads();

  for (int ib = 0; ib < 64; ++ib) {
    const int cur = ib & 1;
    if (ib + 1 < 64) { stageK(cur ^ 1, (ib + 1) * 32); stageV(cur ^ 1, (ib + 1) * 32); }

    v8f st0 = {}, st1 = {};
    {
      v16h k00 = load_frag(&Ktile[cur][0], 64, lane);
      v16h k01 = load_frag(&Ktile[cur][32], 64, lane);
      st0 = WMMA_F32_F16(qa0, k00, st0);
      st0 = WMMA_F32_F16(qa1, k01, st0);
      v16h k10 = load_frag(&Ktile[cur][16 * 64], 64, lane);
      v16h k11 = load_frag(&Ktile[cur][16 * 64 + 32], 64, lane);
      st1 = WMMA_F32_F16(qa0, k10, st1);
      st1 = WMMA_F32_F16(qa1, k11, st1);
    }
    float alpha[8];
#pragma unroll
    for (int r = 0; r < 8; ++r) {
      const float s0 = st0[r] * 0.125f;  // 1/sqrt(64)
      const float s1 = st1[r] * 0.125f;
      float mx = fmaxf(s0, s1);
#pragma unroll
      for (int off = 1; off < 16; off <<= 1)
        mx = fmaxf(mx, __shfl_xor(mx, off, 32));
      const float mnew = fmaxf(mrow[r], mx);
      alpha[r] = __expf(mrow[r] - mnew);
      mrow[r] = mnew;
      const float p0 = __expf(s0 - mnew);
      const float p1 = __expf(s1 - mnew);
      const int rowoff = ((g << 3) + r) * 32;   // C/D layout row -> LDS row
      pw[rowoff + cn]      = (_Float16)p0;
      pw[rowoff + 16 + cn] = (_Float16)p1;
      float rs = p0 + p1;
#pragma unroll
      for (int off = 1; off < 16; off <<= 1)
        rs += __shfl_xor(rs, off, 32);
      lrow[r] = lrow[r] * alpha[r] + rs;
    }
#pragma unroll
    for (int i = 0; i < 4; ++i)
#pragma unroll
      for (int r = 0; r < 8; ++r) o[i][r] *= alpha[r];

    const v16h pa = load_frag(pw, 32, lane);  // P tile as A fragment (16x32)
#pragma unroll
    for (int i = 0; i < 4; ++i) {
      v16h vb = load_frag(&Vtile[cur][(i * 16) * 32], 32, lane);
      o[i] = WMMA_F32_F16(pa, vb, o[i]);
    }

    async_join();
    __syncthreads();
  }

  const int b = bh >> 4, h = bh & 15;
#pragma unroll
  for (int i = 0; i < 4; ++i)
#pragma unroll
    for (int r = 0; r < 8; ++r) {
      const int s = qblk * 128 + wave * 16 + (g << 3) + r;
      const int col = h * 64 + i * 16 + cn;
      attn[((long)b * 2048 + s) * 1024 + col] = (_Float16)(o[i][r] / lrow[r]);
    }
}

// ---------------------------------------------------------------------------
// Output projection: out[8192][1024] = attn[8192][1024] @ WoutT^T + bias (f32)
__global__ __launch_bounds__(256) void out_gemm_kernel(
    const _Float16* __restrict__ A,   // attn [8192][1024]
    const _Float16* __restrict__ BT,  // [1024][1024]
    const float* __restrict__ bias,   // [1024]
    float* __restrict__ out) {
  __shared__ __align__(16) _Float16 Abuf[2][128 * 32];
  __shared__ __align__(16) _Float16 Bbuf[2][128 * 32];
  const int tid = threadIdx.x, lane = tid & 31, wave = tid >> 5;
  const int wm = wave & 3, wn = wave >> 2;
  const long Arow0 = (long)blockIdx.y * 128;
  const long Brow0 = (long)blockIdx.x * 128;

  auto stage = [&](int buf, int k) {
#pragma unroll
    for (int j = 0; j < 2; ++j) {
      const int c = tid + 256 * j;
      const int row = c >> 2, seg = (c & 3) << 3;
      async_copy16(A + (Arow0 + row) * 1024 + k + seg, &Abuf[buf][row * 32 + seg]);
      async_copy16(BT + (Brow0 + row) * 1024 + k + seg, &Bbuf[buf][row * 32 + seg]);
    }
  };

  v8f acc[2][4] = {};
  stage(0, 0);
  async_join();
  __syncthreads();
  for (int ib = 0; ib < 32; ++ib) {
    const int cur = ib & 1;
    if (ib + 1 < 32) stage(cur ^ 1, (ib + 1) * 32);

    v16h a0 = load_frag(&Abuf[cur][(wm * 32) * 32], 32, lane);
    v16h a1 = load_frag(&Abuf[cur][(wm * 32 + 16) * 32], 32, lane);
#pragma unroll
    for (int j = 0; j < 4; ++j) {
      v16h b = load_frag(&Bbuf[cur][(wn * 64 + j * 16) * 32], 32, lane);
      acc[0][j] = WMMA_F32_F16(a0, b, acc[0][j]);
      acc[1][j] = WMMA_F32_F16(a1, b, acc[1][j]);
    }
    async_join();
    __syncthreads();
  }

  const int m0 = (int)Arow0 + wm * 32;
  const int n0 = (int)Brow0 + wn * 64;
  const int g = lane >> 4, cn = lane & 15;
#pragma unroll
  for (int i = 0; i < 2; ++i)
#pragma unroll
    for (int j = 0; j < 4; ++j)
#pragma unroll
      for (int r = 0; r < 8; ++r) {
        const int gm = m0 + i * 16 + (g << 3) + r;
        const int gn = n0 + j * 16 + cn;
        out[(long)gm * 1024 + gn] = acc[i][j][r] + bias[gn];
      }
}

// ---------------------------------------------------------------------------
extern "C" void kernel_launch(void* const* d_in, const int* in_sizes, int n_in,
                              void* d_out, int out_size, void* d_ws, size_t ws_size,
                              hipStream_t stream) {
  const float* x     = (const float*)d_in[0];  // [4,2048,1024]
  const float* w_qkv = (const float*)d_in[1];  // [1024,3072]
  const float* b_qkv = (const float*)d_in[2];  // [3072]
  const float* w_out = (const float*)d_in[3];  // [1024,1024]
  const float* b_out = (const float*)d_in[4];  // [1024]
  float* out = (float*)d_out;                  // [4,2048,1024]

  char* ws = (char*)d_ws;
  _Float16* xh    = (_Float16*)(ws);              // 16.78 MB
  _Float16* wqkvT = (_Float16*)(ws + 16777216);   //  6.29 MB
  _Float16* woutT = (_Float16*)(ws + 23068672);   //  2.10 MB
  _Float16* Qb    = (_Float16*)(ws + 25165824);   // 16.78 MB
  _Float16* Kb    = (_Float16*)(ws + 41943040);   // 16.78 MB
  _Float16* Vt    = (_Float16*)(ws + 58720256);   // 16.78 MB
  _Float16* attn  = (_Float16*)(ws + 75497472);   // 16.78 MB  (total ~92.3 MB)

  cvt_f16_kernel<<<8388608 / 256, 256, 0, stream>>>(x, xh, 8388608);
  transpose_cvt_kernel<<<3145728 / 256, 256, 0, stream>>>(w_qkv, wqkvT, 1024, 3072);
  transpose_cvt_kernel<<<1048576 / 256, 256, 0, stream>>>(w_out, woutT, 1024, 1024);
  qkv_gemm_kernel<<<dim3(24, 64), 256, 0, stream>>>(xh, wqkvT, b_qkv, Qb, Kb, Vt);
  flash_attn_kernel<<<1024, 256, 0, stream>>>(Qb, Kb, Vt, attn);
  out_gemm_kernel<<<dim3(8, 64), 256, 0, stream>>>(attn, woutT, b_out, out);
}
